// GaussianRenderer_13443247637320
// MI455X (gfx1250) — compile-verified, matrix operand order checked by reading, not run on hardware
//
#include <hip/hip_runtime.h>

#define N_GAUSS 512
#define IMG_H 256
#define IMG_W 256
#define REPS 1e-4f

typedef __attribute__((ext_vector_type(16))) _Float16 v16h;
typedef __attribute__((ext_vector_type(8)))  float    v8f;

#if __has_builtin(__builtin_amdgcn_exp2f)
#define EXP2F(x) __builtin_amdgcn_exp2f(x)
#else
#define EXP2F(x) exp2f(x)
#endif

// Workspace layout:
//   [0, 16384)        : sorted per-gaussian params, 512 x 8 floats
//                       {mx, my, A2, B2, C2, coeff, 0, 0}  (A2/B2/C2 prescaled by 0.5*log2e)
//   [16384, 32768)    : B fragments: 16 batches x 32 lanes x 16 halfs
//                       (colors in v_wmma_f32_16x16x32_f16 B-matrix VGPR layout, cols>=3 zero)

__global__ __launch_bounds__(N_GAUSS) void prep_kernel(
    const float* __restrict__ means3D,
    const float* __restrict__ covs3d,
    const float* __restrict__ colors,
    const float* __restrict__ opac,
    const float* __restrict__ Km,
    const float* __restrict__ Rm,
    const float* __restrict__ tv,
    float* __restrict__ wsP,
    _Float16* __restrict__ wsB)
{
  __shared__ float dS[N_GAUSS];
  __shared__ float colS[N_GAUSS][3];
  const int g = threadIdx.x;

  // camera constants (uniform -> scalar loads)
  const float K00=Km[0],K01=Km[1],K02=Km[2];
  const float K10=Km[3],K11=Km[4],K12=Km[5];
  const float K20=Km[6],K21=Km[7],K22=Km[8];
  float R[9];
  #pragma unroll
  for (int i=0;i<9;i++) R[i]=Rm[i];
  const float t0=tv[0], t1=tv[1], t2=tv[2];

  // --- projection ---
  const float m0=means3D[g*3+0], m1=means3D[g*3+1], m2=means3D[g*3+2];
  const float camx = R[0]*m0 + R[1]*m1 + R[2]*m2 + t0;
  const float camy = R[3]*m0 + R[4]*m1 + R[5]*m2 + t1;
  const float camz = R[6]*m0 + R[7]*m1 + R[8]*m2 + t2;
  const float depth = fmaxf(camz, 1.0f);
  const float sx = K00*camx + K01*camy + K02*camz;
  const float sy = K10*camx + K11*camy + K12*camz;
  const float sz = K20*camx + K21*camy + K22*camz;
  const float mx = sx / sz, my = sy / sz;
  const float zi = 1.0f / camz, zi2 = zi*zi;
  const float J00 = K00*zi, J01 = K01*zi, J02 = -(K00*camx + K01*camy)*zi2;
  const float J10 = 0.0f,   J11 = K11*zi, J12 = -(K11*camy)*zi2;

  // --- covs_cam = R C R^T ; covs2D = J covs_cam J^T ---
  float C[9];
  #pragma unroll
  for (int i=0;i<9;i++) C[i]=covs3d[g*9+i];
  float M[9];
  #pragma unroll
  for (int i=0;i<3;i++)
    #pragma unroll
    for (int j=0;j<3;j++)
      M[i*3+j] = R[i*3+0]*C[j] + R[i*3+1]*C[3+j] + R[i*3+2]*C[6+j];
  float Cc[9];
  #pragma unroll
  for (int i=0;i<3;i++)
    #pragma unroll
    for (int j=0;j<3;j++)
      Cc[i*3+j] = M[i*3+0]*R[j*3+0] + M[i*3+1]*R[j*3+1] + M[i*3+2]*R[j*3+2];

  const float W0 = J00*Cc[0] + J01*Cc[3] + J02*Cc[6];
  const float W1 = J00*Cc[1] + J01*Cc[4] + J02*Cc[7];
  const float W2 = J00*Cc[2] + J01*Cc[5] + J02*Cc[8];
  const float X0 = J10*Cc[0] + J11*Cc[3] + J12*Cc[6];
  const float X1 = J10*Cc[1] + J11*Cc[4] + J12*Cc[7];
  const float X2 = J10*Cc[2] + J11*Cc[5] + J12*Cc[8];
  const float v00 = W0*J00 + W1*J01 + W2*J02;
  const float v01 = W0*J10 + W1*J11 + W2*J12;
  const float v10 = X0*J00 + X1*J01 + X2*J02;
  const float v11 = X0*J10 + X1*J11 + X2*J12;

  // cov = covs2D + EPS*I ; exact inverse (matches jnp.linalg.inv); det+EPS for norm
  const float c00 = v00 + REPS, c01 = v01, c10 = v10, c11 = v11 + REPS;
  const float detn = c00*c11 - c01*c10;
  const float rdet = 1.0f / detn;
  const float a  =  c11*rdet;
  const float bs = -(c01 + c10)*rdet;   // inv01 + inv10
  const float cc =  c00*rdet;
  const float det = detn + REPS;
  const float norm = 0.15915494309189535f * rsqrtf(det); // 1/(2*pi*sqrt(det))
  const bool valid = (depth > 1.0f) && (depth < 50.0f);
  const float coeff = valid ? opac[g]*norm : 0.0f;
  const float HL2E = 0.72134752044448170368f;            // 0.5*log2(e)
  const float A2 = a*HL2E, B2 = bs*HL2E, C2 = cc*HL2E;

  // --- stable depth sort via rank counting ---
  dS[g] = depth;
  __syncthreads();
  int rank = 0;
  for (int j=0;j<N_GAUSS;j++){
    const float dj = dS[j];
    rank += (dj < depth || (dj == depth && j < g)) ? 1 : 0;
  }
  float* P = wsP + rank*8;
  P[0]=mx; P[1]=my; P[2]=A2; P[3]=B2; P[4]=C2; P[5]=coeff; P[6]=0.0f; P[7]=0.0f;
  colS[rank][0] = colors[g*3+0];
  colS[rank][1] = colors[g*3+1];
  colS[rank][2] = colors[g*3+2];
  __syncthreads();

  // --- build B fragments (16-bit B 32x16 layout: lanes 0-15 col n, K={0-7,16-23};
  //     lanes 16-31 col n, K={8-15,24-31}; 2 halfs per VGPR) ---
  const int batch = g >> 5, lane = g & 31;
  const int n = lane & 15;
  const int kb = (lane < 16) ? 0 : 8;
  union { _Float16 h[16]; uint4 u[2]; } tmp;
  #pragma unroll
  for (int i=0;i<16;i++){
    const int kl = kb + ((i < 8) ? i : (i + 8));
    const int k  = (batch << 5) + kl;
    const float v = (n < 3) ? colS[k][n] : 0.0f;
    tmp.h[i] = (_Float16)v;
  }
  uint4* du = (uint4*)(wsB + (g << 4));
  du[0] = tmp.u[0];
  du[1] = tmp.u[1];
}

__global__ __launch_bounds__(IMG_W) void render_kernel(
    const float* __restrict__ wsP,
    const _Float16* __restrict__ wsB,
    float* __restrict__ out)
{
  __shared__ alignas(16) float    pS[N_GAUSS*8];        // 16 KB param cache
  __shared__ alignas(16) _Float16 wbuf[8][32][32];      // 16 KB weight staging (per wave 32px x 32k)
  __shared__ float ostage[8][32][4];                    // 4 KB RGB routing

  const int wave = threadIdx.x >> 5;
  const int lane = threadIdx.x & 31;
  const int py = blockIdx.x;
  const int px = threadIdx.x;           // one image row per block

  __builtin_prefetch(wsP, 0, 1);        // global_prefetch_b8
  __builtin_prefetch(wsB, 0, 1);

  // cache sorted params in LDS
  {
    const uint4* src = (const uint4*)wsP;
    uint4* dst = (uint4*)pS;
    for (int i = threadIdx.x; i < N_GAUSS*2; i += IMG_W) dst[i] = src[i];
  }
  __syncthreads();

  const float fx = (float)px, fy = (float)py;
  float T = 1.0f;
  v8f acc0 = {};   // pixels wavebase+0..15  (16x16 f32 D)
  v8f acc1 = {};   // pixels wavebase+16..31

  for (int b = 0; b < 16; ++b) {
    // sequential transparency cumprod over this 32-gaussian batch (sorted order)
    #pragma unroll
    for (int j = 0; j < 32; ++j) {
      const float* p = pS + (((b << 5) + j) << 3);
      const float dx = fx - p[0];
      const float dy = fy - p[1];
      const float e  = -(p[2]*dx*dx + p[3]*dx*dy + p[4]*dy*dy);
      const float alpha = p[5] * EXP2F(e);
      T *= (1.0f - alpha);
      wbuf[wave][lane][j] = (_Float16)(alpha * T);   // weights = alpha * cumprod(1-alpha) incl. self
    }
    __builtin_amdgcn_wave_barrier();

    // gather A fragments in 16-bit A 16x32 layout:
    //   lanes 0-15: row M=lane, halfs = K{0-7,16-23}; lanes 16-31: row M=lane-16, K{8-15,24-31}
    const int p0  = lane & 15;
    const int off = (lane < 16) ? 0 : 8;
    const uint4* r0 = (const uint4*)&wbuf[wave][p0][off];
    const uint4* r1 = (const uint4*)&wbuf[wave][p0 + 16][off];
    union Fr { uint4 u[2]; v16h h; };
    Fr a0, a1, bf;
    a0.u[0] = r0[0]; a0.u[1] = r0[2];
    a1.u[0] = r1[0]; a1.u[1] = r1[2];
    const uint4* bp = (const uint4*)(wsB + (((b << 5) + lane) << 4));
    bf.u[0] = bp[0]; bf.u[1] = bp[1];

    acc0 = __builtin_amdgcn_wmma_f32_16x16x32_f16(false, a0.h, false, bf.h,
                                                  (short)0, acc0, false, false);
    acc1 = __builtin_amdgcn_wmma_f32_16x16x32_f16(false, a1.h, false, bf.h,
                                                  (short)0, acc1, false, false);
    __builtin_amdgcn_wave_barrier();
  }

  // route D (VGPR r: lanes 0-15 -> M=r,N=lane; lanes 16-31 -> M=8+r,N=lane-16) to pixels
  const int n = lane & 15;
  const int mbase = (lane < 16) ? 0 : 8;
  if (n < 3) {
    #pragma unroll
    for (int r = 0; r < 8; ++r) {
      ostage[wave][mbase + r][n]      = acc0[r];
      ostage[wave][16 + mbase + r][n] = acc1[r];
    }
  }
  __builtin_amdgcn_wave_barrier();

  float* o = out + (py * IMG_W + px) * 3;
  o[0] = ostage[wave][lane][0];
  o[1] = ostage[wave][lane][1];
  o[2] = ostage[wave][lane][2];
}

extern "C" void kernel_launch(void* const* d_in, const int* in_sizes, int n_in,
                              void* d_out, int out_size, void* d_ws, size_t ws_size,
                              hipStream_t stream) {
  const float* means3D = (const float*)d_in[0];
  const float* covs3d  = (const float*)d_in[1];
  const float* colors  = (const float*)d_in[2];
  const float* opac    = (const float*)d_in[3];
  const float* Km      = (const float*)d_in[4];
  const float* Rm      = (const float*)d_in[5];
  const float* tv      = (const float*)d_in[6];
  // d_in[7], d_in[8] are H, W (fixed at 256 here)

  float*    wsP = (float*)d_ws;
  _Float16* wsB = (_Float16*)((char*)d_ws + N_GAUSS * 8 * sizeof(float));

  prep_kernel<<<1, N_GAUSS, 0, stream>>>(means3D, covs3d, colors, opac, Km, Rm, tv, wsP, wsB);
  render_kernel<<<IMG_H, IMG_W, 0, stream>>>(wsP, wsB, (float*)d_out);
}
